// NoisyTopKRouter_19464791786099
// MI455X (gfx1250) — matrix operand order repaired and checked by reading, not compiled, not run on hardware
//
#include <hip/hip_runtime.h>
#include <hip/hip_bf16.h>
#include <math.h>

#define TOKENS    16384
#define N_EMBD    4096
#define N_EXPERTS 64

typedef __attribute__((ext_vector_type(16))) __bf16         v16bf;
typedef __attribute__((ext_vector_type(16))) unsigned short v16u;
typedef __attribute__((ext_vector_type(8)))  float          v8f;

// Round-to-nearest-even fp32 -> bf16 (bit pattern).
__device__ __forceinline__ unsigned short f32_to_bf16(float f) {
    unsigned int u = __float_as_uint(f);
    u += 0x7FFFu + ((u >> 16) & 1u);
    return (unsigned short)(u >> 16);
}

__device__ __forceinline__ v16bf pack_a(float4 f0, float4 f1, float4 f2, float4 f3) {
    v16u au;
    au[0]  = f32_to_bf16(f0.x); au[1]  = f32_to_bf16(f0.y);
    au[2]  = f32_to_bf16(f0.z); au[3]  = f32_to_bf16(f0.w);
    au[4]  = f32_to_bf16(f1.x); au[5]  = f32_to_bf16(f1.y);
    au[6]  = f32_to_bf16(f1.z); au[7]  = f32_to_bf16(f1.w);
    au[8]  = f32_to_bf16(f2.x); au[9]  = f32_to_bf16(f2.y);
    au[10] = f32_to_bf16(f2.z); au[11] = f32_to_bf16(f2.w);
    au[12] = f32_to_bf16(f3.x); au[13] = f32_to_bf16(f3.y);
    au[14] = f32_to_bf16(f3.z); au[15] = f32_to_bf16(f3.w);
    return __builtin_bit_cast(v16bf, au);
}

// ---------------------------------------------------------------------------
// Kernel 1: convert route_w (fp32 [64][4096]) -> bf16 in workspace.
// ---------------------------------------------------------------------------
__global__ __launch_bounds__(256) void moe_convert_w(
    const float* __restrict__ w, unsigned short* __restrict__ wb, int n) {
    int base = (blockIdx.x * blockDim.x + threadIdx.x) * 4;
    if (base < n) {
        float4 f = *(const float4*)(w + base);
        unsigned int lo = (unsigned int)f32_to_bf16(f.x) |
                          ((unsigned int)f32_to_bf16(f.y) << 16);
        unsigned int hi = (unsigned int)f32_to_bf16(f.z) |
                          ((unsigned int)f32_to_bf16(f.w) << 16);
        *(uint2*)(wb + base) = make_uint2(lo, hi);
    }
}

// ---------------------------------------------------------------------------
// Kernel 2: logits GEMM (bf16 WMMA, f32 acc) + bias + top-2 + sparse softmax.
// Block = 128 threads = 4 waves; each wave computes 16 tokens x 64 experts.
// K loop is software-pipelined: A (HBM stream) loads run one step ahead,
// all 4 B fragments live in distinct registers so their 8 loads clause
// together and the 4 WMMAs issue back-to-back behind one wait.
// ---------------------------------------------------------------------------
__global__ __launch_bounds__(128) void moe_router(
    const float*          __restrict__ x,       // [TOKENS][N_EMBD] fp32
    const unsigned short* __restrict__ wbf,     // [64][N_EMBD] bf16 bits
    const float*          __restrict__ route_b, // [64]
    float*                __restrict__ out)     // [TOKENS][64] f32, then [TOKENS][2] i32
{
    __shared__ float lds_logits[64][65];   // 64 tokens/block, padded rows
    __shared__ float lds_bias[N_EXPERTS];

    const int tid  = threadIdx.x;
    const int wave = tid >> 5;          // 0..3
    const int lane = tid & 31;
    const int hl   = lane >> 4;         // lane half (0/1)
    const int ln   = lane & 15;         // A: row within tile / B: column

    if (tid < N_EXPERTS) lds_bias[tid] = route_b[tid];

    const int tokBlock = blockIdx.x * 64;          // 256 blocks * 64 tokens
    const int tokWave  = tokBlock + wave * 16;

    // ---- Phase 1: GEMM via v_wmma_f32_16x16x32_bf16 -----------------------
    // A fragment element j: K = hl*16 + j.
    const float* xrow = x + (size_t)(tokWave + ln) * N_EMBD + hl * 16;
    // B fragment: lane ln = expert column, element j: K = hl*16 + j.
    const unsigned short* w0 = wbf + (size_t)( 0 + ln) * N_EMBD + hl * 16;
    const unsigned short* w1 = wbf + (size_t)(16 + ln) * N_EMBD + hl * 16;
    const unsigned short* w2 = wbf + (size_t)(32 + ln) * N_EMBD + hl * 16;
    const unsigned short* w3 = wbf + (size_t)(48 + ln) * N_EMBD + hl * 16;

    v8f acc0 = {}, acc1 = {}, acc2 = {}, acc3 = {};

    // Prologue: A floats for k0 = 0.
    float4 f0 = *(const float4*)(xrow + 0);
    float4 f1 = *(const float4*)(xrow + 4);
    float4 f2 = *(const float4*)(xrow + 8);
    float4 f3 = *(const float4*)(xrow + 12);

    for (int k0 = 0; k0 < N_EMBD; k0 += 32) {
        // B fragments for this step: 4 distinct register sets, 8 clause'd loads.
        v16bf b0 = *(const v16bf*)(w0 + k0);
        v16bf b1 = *(const v16bf*)(w1 + k0);
        v16bf b2 = *(const v16bf*)(w2 + k0);
        v16bf b3 = *(const v16bf*)(w3 + k0);

        // A loads for the NEXT step (in-bounds dummy re-read on last step).
        const int kn = (k0 + 32 < N_EMBD) ? (k0 + 32) : k0;
        float4 g0 = *(const float4*)(xrow + kn + 0);
        float4 g1 = *(const float4*)(xrow + kn + 4);
        float4 g2 = *(const float4*)(xrow + kn + 8);
        float4 g3 = *(const float4*)(xrow + kn + 12);

        // Convert current A chunk (VALU overlaps the loads above).
        v16bf a = pack_a(f0, f1, f2, f3);

        acc0 = __builtin_amdgcn_wmma_f32_16x16x32_bf16(
            false, a, false, b0, (short)0, acc0, false, false);
        acc1 = __builtin_amdgcn_wmma_f32_16x16x32_bf16(
            false, a, false, b1, (short)0, acc1, false, false);
        acc2 = __builtin_amdgcn_wmma_f32_16x16x32_bf16(
            false, a, false, b2, (short)0, acc2, false, false);
        acc3 = __builtin_amdgcn_wmma_f32_16x16x32_bf16(
            false, a, false, b3, (short)0, acc3, false, false);

        f0 = g0; f1 = g1; f2 = g2; f3 = g3;
    }

    // C layout: VGPR i, lanes 0-15 -> M=i, N=ln; lanes 16-31 -> M=8+i, N=ln.
    const int rowBase = wave * 16 + hl * 8;
#pragma unroll
    for (int i = 0; i < 8; ++i) {
        lds_logits[rowBase + i][ 0 + ln] = acc0[i];
        lds_logits[rowBase + i][16 + ln] = acc1[i];
        lds_logits[rowBase + i][32 + ln] = acc2[i];
        lds_logits[rowBase + i][48 + ln] = acc3[i];
    }

    __syncthreads();

    // ---- Phase 2: bias + top-2 + sparse softmax ---------------------------
    if (tid < 64) {
        float best0 = -1e30f, best1 = -1e30f;
        int   idx0  = 0,      idx1  = 0;
        for (int e = 0; e < N_EXPERTS; ++e) {
            float v = lds_logits[tid][e] + lds_bias[e];
            if (v > best0) {            // strict '>' => lowest index wins ties
                best1 = best0; idx1 = idx0;
                best0 = v;     idx0 = e;
            } else if (v > best1) {
                best1 = v;     idx1 = e;
            }
        }
        // softmax over {best0, best1} with -inf elsewhere
        float e1  = __expf(best1 - best0);
        float inv = 1.0f / (1.0f + e1);
        float p0  = inv, p1 = e1 * inv;

        const size_t gtok = (size_t)tokBlock + tid;
        float* orow = out + gtok * N_EXPERTS;
#pragma unroll
        for (int i = 0; i < 16; ++i) {
            float4 z = make_float4(0.f, 0.f, 0.f, 0.f);
            if ((idx0 >> 2) == i) ((float*)&z)[idx0 & 3] = p0;
            if ((idx1 >> 2) == i) ((float*)&z)[idx1 & 3] = p1;
            *(float4*)(orow + i * 4) = z;
        }
        int* oidx = (int*)(out + (size_t)TOKENS * N_EXPERTS);
        oidx[gtok * 2 + 0] = idx0;
        oidx[gtok * 2 + 1] = idx1;
    }
}

// ---------------------------------------------------------------------------
extern "C" void kernel_launch(void* const* d_in, const int* in_sizes, int n_in,
                              void* d_out, int out_size, void* d_ws, size_t ws_size,
                              hipStream_t stream) {
    const float* x       = (const float*)d_in[0];
    // d_in[1] (noise) is dead code in the reference output — skipped.
    const float* route_w = (const float*)d_in[2];
    const float* route_b = (const float*)d_in[3];
    // d_in[4], d_in[5] (noise_w, noise_b) also dead — skipped.

    unsigned short* wbf = (unsigned short*)d_ws;   // 64*4096*2 = 512 KB
    float* out = (float*)d_out;

    const int nW = N_EXPERTS * N_EMBD;
    moe_convert_w<<<nW / (256 * 4), 256, 0, stream>>>(route_w, wbf, nW);
    moe_router<<<TOKENS / 64, 128, 0, stream>>>(x, wbf, route_b, out);
}
